// MessagePassingLayer_7971459301457
// MI455X (gfx1250) — matrix operand order repaired and checked by reading, not compiled, not run on hardware
//
#include <hip/hip_runtime.h>
#include <hip/hip_bf16.h>

typedef __attribute__((ext_vector_type(16))) _Float16 v16h;
typedef __attribute__((ext_vector_type(8)))  float    v8f;
typedef __attribute__((ext_vector_type(4)))  int      v4i;

#define AS_GLOBAL __attribute__((address_space(1)))
#define AS_LOCAL  __attribute__((address_space(3)))

#define IN_DIM  64
#define K_DIM   128   // 2*IN_DIM (concat of two 64-dim features)
#define OUT_DIM 64
#define TILE    128   // rows (edges or nodes) per block
#define THREADS 256   // 8 wave32 waves -> 8 M-tiles of 16 rows

#if defined(__AMDGCN__) && __has_builtin(__builtin_amdgcn_global_load_async_to_lds_b128)
#define USE_ASYNC_LDS 1
#else
#define USE_ASYNC_LDS 0
#endif

// ---- copy one 64-float feature row (256B) into LDS ----
__device__ __forceinline__ void gather_row(const float* __restrict__ src,
                                           float* __restrict__ dst) {
#if USE_ASYNC_LDS
#pragma unroll
  for (int i = 0; i < 16; ++i) {
    __builtin_amdgcn_global_load_async_to_lds_b128(
        (AS_GLOBAL v4i*)(src + i * 4),
        (AS_LOCAL  v4i*)(dst + i * 4), 0, 0);
  }
#else
#pragma unroll
  for (int i = 0; i < 16; ++i) {
    float4 v = ((const float4*)src)[i];
    ((float4*)dst)[i] = v;
  }
#endif
}

__device__ __forceinline__ void wait_gather() {
#if USE_ASYNC_LDS
  asm volatile("s_wait_asynccnt 0" ::: "memory");
#endif
}

// ---- WMMA A-fragment from f32 LDS row (layout per CDNA5 ISA §7.12.2) ----
// A (16x32 f16): lane = {m=lane&15, khalf=lane>>4}; elements 0..7 hold
// K = kk*32 + khalf*8 .. +7, elements 8..15 hold K = kk*32 + 16 + khalf*8 .. +7.
__device__ __forceinline__ v16h load_a_frag(const float* __restrict__ Arow,
                                            int kk, int khalf, float scale) {
  const float* p0 = Arow + kk * 32 + khalf * 8;
  v8f lo = *(const v8f*)p0;         // 32B LDS read
  v8f hi = *(const v8f*)(p0 + 16);  // 32B LDS read
  v16h r;
#pragma unroll
  for (int i = 0; i < 8; ++i) {
    r[i]     = (_Float16)(lo[i] * scale);
    r[i + 8] = (_Float16)(hi[i] * scale);
  }
  return r;
}

// ------------------------- zero workspace -------------------------
__global__ void gnn_zero_kernel(float* __restrict__ p, size_t n) {
  size_t i = (size_t)blockIdx.x * blockDim.x + threadIdx.x;
  size_t stride = (size_t)gridDim.x * blockDim.x;
  for (; i < n; i += stride) p[i] = 0.0f;
}

// ------------------------- pack W into per-lane B-fragment image ----------
// Bimg[(nt*4+kk)*32 + lane] is the v16h fragment lane `lane` needs for
// N-tile nt, K-step kk: column n = nt*16 + (lane&15),
// K = kk*32 + (lane>>4)*8 + i (i<8), kk*32 + 16 + (lane>>4)*8 + (i-8) (i>=8).
__global__ void gnn_pack_w_kernel(const float* __restrict__ W,   // [128][64]
                                  _Float16* __restrict__ Bimg) { // 512*16 halfs
  int t = blockIdx.x * blockDim.x + threadIdx.x;
  if (t >= 512) return;
  const int lane = t & 31;
  const int kk   = (t >> 5) & 3;
  const int nt   = t >> 7;
  const int n     = nt * 16 + (lane & 15);
  const int kbase = kk * 32 + (lane >> 4) * 8;
  _Float16* dst = Bimg + (size_t)t * 16;
#pragma unroll
  for (int i = 0; i < 16; ++i) dst[i] = (_Float16)W[(kbase + i + (i >= 8 ? 8 : 0)) * OUT_DIM + n];
}

// ------------------------- edge (message) kernel -------------------------
__global__ __launch_bounds__(THREADS)
void gnn_edge_kernel(const float* __restrict__ x,
                     const int*   __restrict__ ei,      // [2][nEdges]
                     const _Float16* __restrict__ Bimg, // packed W_msg frags
                     const float* __restrict__ bm,      // [64]
                     float* __restrict__ agg,           // [nNodes][64]
                     float* __restrict__ cnt,           // [nNodes]
                     int nEdges) {
  __shared__ __align__(16) float lds_A[TILE * K_DIM];  // 64 KB, f32 tile
  __shared__ int lds_tgt[TILE];

  // ---- gather phase: one (edge,half) pair per thread ----
  const int t    = threadIdx.x;
  const int erow = t >> 1;   // 0..127
  const int h    = t & 1;    // 0 = src half, 1 = tgt half
  const int e    = blockIdx.x * TILE + erow;
  int nodeIdx = -1;
  if (e < nEdges) nodeIdx = ei[h * nEdges + e];

  float* dst = &lds_A[erow * K_DIM + h * IN_DIM];
  if (nodeIdx >= 0) {
    gather_row(x + (size_t)nodeIdx * IN_DIM, dst);
  } else {
#pragma unroll
    for (int i = 0; i < IN_DIM; ++i) dst[i] = 0.0f;
  }
  if (h) {
    lds_tgt[erow] = nodeIdx;                       // -1 marks invalid row
    if (nodeIdx >= 0) atomicAdd(&cnt[nodeIdx], 1.0f);
  }
  wait_gather();
  __syncthreads();

  // ---- WMMA phase: wave w owns rows [w*16, w*16+16) ----
  const int lane  = t & 31;
  const int mt    = t >> 5;
  const int lrow  = lane & 15;
  const int khalf = lane >> 4;

  v8f acc[4] = {};  // 4 N-tiles of 16 cols
  const float* Arow = &lds_A[(mt * 16 + lrow) * K_DIM];
  const v16h*  Bfr  = (const v16h*)Bimg;
#pragma unroll
  for (int kk = 0; kk < 4; ++kk) {
    v16h a = load_a_frag(Arow, kk, khalf, 1.0f);
#pragma unroll
    for (int nt = 0; nt < 4; ++nt) {
      v16h b = Bfr[(nt * 4 + kk) * 32 + lane];   // 32B from WGP$-resident image
      acc[nt] = __builtin_amdgcn_wmma_f32_16x16x32_f16(
          false, a, false, b, (short)0, acc[nt], false, false);
    }
  }

  // ---- bias + relu + scatter-add (L2-resident accumulator) ----
#pragma unroll
  for (int nt = 0; nt < 4; ++nt) {
    const int n = nt * 16 + lrow;
    const float bias = bm[n];
#pragma unroll
    for (int r = 0; r < 8; ++r) {
      const int mloc = mt * 16 + khalf * 8 + r;  // C layout: vgpr r -> M=r(+8)
      const int tgt  = lds_tgt[mloc];
      float v = acc[nt][r] + bias;
      v = v > 0.0f ? v : 0.0f;
      if (tgt >= 0) atomicAdd(&agg[(size_t)tgt * OUT_DIM + n], v);
    }
  }
}

// ------------------------- node (update) kernel -------------------------
__global__ __launch_bounds__(THREADS)
void gnn_node_kernel(const float* __restrict__ x,
                     const float* __restrict__ agg,
                     const float* __restrict__ cnt,
                     const _Float16* __restrict__ Bimg, // packed W_upd frags
                     const float* __restrict__ bu,      // [64]
                     float* __restrict__ out,           // [nNodes][64]
                     int nNodes) {
  __shared__ __align__(16) float lds_A[TILE * K_DIM];  // 64 KB
  __shared__ float lds_inv[TILE];                      // 1/max(cnt,1) per row

  const int t    = threadIdx.x;
  const int nrow = t >> 1;
  const int h    = t & 1;     // 0 = x half, 1 = summed-message half
  const int node = blockIdx.x * TILE + nrow;

  float* dst = &lds_A[nrow * K_DIM + h * IN_DIM];
  if (node < nNodes) {
    const float* src = (h == 0) ? (x + (size_t)node * IN_DIM)
                                : (agg + (size_t)node * OUT_DIM);
    gather_row(src, dst);
    if (h) {
      const float c = cnt[node];
      lds_inv[nrow] = 1.0f / (c > 1.0f ? c : 1.0f);
    }
  } else {
#pragma unroll
    for (int i = 0; i < IN_DIM; ++i) dst[i] = 0.0f;
    if (h) lds_inv[nrow] = 1.0f;
  }
  wait_gather();
  __syncthreads();

  const int lane  = t & 31;
  const int mt    = t >> 5;
  const int lrow  = lane & 15;
  const int khalf = lane >> 4;

  v8f acc[4] = {};
  const float* Arow = &lds_A[(mt * 16 + lrow) * K_DIM];
  const float  inv  = lds_inv[mt * 16 + lrow];
  const v16h*  Bfr  = (const v16h*)Bimg;
#pragma unroll
  for (int kk = 0; kk < 4; ++kk) {
    // kk 0..1 = raw x (K 0..63); kk 2..3 = aggregated sum -> scale to mean.
    const float scale = (kk >= 2) ? inv : 1.0f;
    v16h a = load_a_frag(Arow, kk, khalf, scale);
#pragma unroll
    for (int nt = 0; nt < 4; ++nt) {
      v16h b = Bfr[(nt * 4 + kk) * 32 + lane];
      acc[nt] = __builtin_amdgcn_wmma_f32_16x16x32_f16(
          false, a, false, b, (short)0, acc[nt], false, false);
    }
  }

#pragma unroll
  for (int nt = 0; nt < 4; ++nt) {
    const int n = nt * 16 + lrow;
    const float bias = bu[n];
#pragma unroll
    for (int r = 0; r < 8; ++r) {
      const int mloc = mt * 16 + khalf * 8 + r;
      const int nd   = blockIdx.x * TILE + mloc;
      float v = acc[nt][r] + bias;
      v = v > 0.0f ? v : 0.0f;
      if (nd < nNodes) out[(size_t)nd * OUT_DIM + n] = v;
    }
  }
}

// ------------------------- launcher -------------------------
extern "C" void kernel_launch(void* const* d_in, const int* in_sizes, int n_in,
                              void* d_out, int out_size, void* d_ws, size_t ws_size,
                              hipStream_t stream) {
  const float* x  = (const float*)d_in[0];
  const int*   ei = (const int*)d_in[1];   // [2][E] (int32 on device)
  const float* Wm = (const float*)d_in[2];
  const float* bm = (const float*)d_in[3];
  const float* Wu = (const float*)d_in[4];
  const float* bu = (const float*)d_in[5];

  const int nNodes = in_sizes[0] / IN_DIM;
  const int nEdges = in_sizes[1] / 2;

  // Workspace layout: agg [N][64] f32 | cnt [N] f32 | BimgM 16KB | BimgU 16KB
  const size_t aggElems = (size_t)nNodes * OUT_DIM;
  float* agg = (float*)d_ws;
  float* cnt = agg + aggElems;
  size_t byteOff = (aggElems + (size_t)nNodes) * sizeof(float);
  byteOff = (byteOff + 255) & ~(size_t)255;
  _Float16* BimgM = (_Float16*)((char*)d_ws + byteOff);
  _Float16* BimgU = BimgM + 512 * 16;

  gnn_zero_kernel<<<2048, 256, 0, stream>>>(agg, aggElems + (size_t)nNodes);
  gnn_pack_w_kernel<<<2, 256, 0, stream>>>(Wm, BimgM);
  gnn_pack_w_kernel<<<2, 256, 0, stream>>>(Wu, BimgU);

  const int eblocks = (nEdges + TILE - 1) / TILE;
  gnn_edge_kernel<<<eblocks, THREADS, 0, stream>>>(x, ei, BimgM, bm, agg, cnt,
                                                   nEdges);

  const int nblocks = (nNodes + TILE - 1) / TILE;
  gnn_node_kernel<<<nblocks, THREADS, 0, stream>>>(x, agg, cnt, BimgU, bu,
                                                   (float*)d_out, nNodes);
}
